// ProtoSeg_58832462021234
// MI455X (gfx1250) — matrix operand order repaired and checked by reading, not compiled, not run on hardware
//
#include <hip/hip_runtime.h>
#include <hip/hip_bf16.h>

typedef __attribute__((ext_vector_type(16))) _Float16 v16h;
typedef __attribute__((ext_vector_type(8)))  float    v8f;

#define HW     16384          // 128*128
#define NPIX   (25*16384)     // output pixels per batch
#define PW     130            // padded width/height (1-px zero halo)
#define PHW    (130*130)      // 16900 padded pixels per image
#define NPAD   (25*PHW)       // padded pixels per batch

// map unpadded pixel index -> padded pixel index
__device__ __forceinline__ long long pad_index(long long p) {
    long long n   = p >> 14;
    long long rem = p & 16383;
    return n * PHW + ((rem >> 7) + 1) * PW + (rem & 127) + 1;
}

// ---------------------------------------------------------------------------
// Pack NCHW f32 input -> padded [n][130][130][Cpad] f16 (halo + tail = 0)
// ---------------------------------------------------------------------------
__global__ void pack_input(const float* __restrict__ img, _Float16* __restrict__ out,
                           int Ci, int Cpad) {
    long long idx = (long long)blockIdx.x * blockDim.x + threadIdx.x;
    long long total = (long long)NPAD * Cpad;
    if (idx >= total) return;
    int c = (int)(idx % Cpad);
    long long p  = idx / Cpad;
    long long n  = p / PHW;
    int pp = (int)(p % PHW);
    int py = pp / PW, px = pp % PW;
    float v = 0.f;
    if (c < Ci && py >= 1 && py <= 128 && px >= 1 && px <= 128) {
        long long yx = (long long)(py - 1) * 128 + (px - 1);
        v = img[(n * Ci + c) * HW + yx];
    }
    out[idx] = (_Float16)v;
}

// ---------------------------------------------------------------------------
// Repack OIHW f32 weights into WMMA-B fragment order (f16):
// idx = (((tap*ksteps+s)*4 + ntile)*32 + lane)*16 + e; krow = 16*(lane>>4)+e, ncol = lane&15
// ---------------------------------------------------------------------------
__global__ void pack_weights(const float* __restrict__ w, _Float16* __restrict__ Bp,
                             int Ci, int Cpad) {
    int ksteps = Cpad >> 5;
    int total  = 9 * ksteps * 4 * 512;
    int idx = blockIdx.x * blockDim.x + threadIdx.x;
    if (idx >= total) return;
    int i    = (idx >> 4) & 15;
    int krow = (((idx >> 8) & 1) << 4) + (idx & 15);
    int rest = idx >> 9;
    int ntile = rest & 3;
    int ts    = rest >> 2;
    int s   = ts % ksteps;
    int tap = ts / ksteps;
    int ci = s * 32 + krow;
    int co = ntile * 16 + i;
    float val = 0.f;
    if (ci < Ci) val = w[(co * Ci + ci) * 9 + tap];
    Bp[idx] = (_Float16)val;
}

// ---------------------------------------------------------------------------
// Implicit-GEMM conv3x3 SAME via v_wmma_f32_16x16x32_f16.
// Input is halo-padded NHWC-f16, so every tap load is unconditionally valid:
// no EXEC predication, no zero-fill in the k-loop.
// Block = 256 threads = 8 wave32s; tile = 128 pixels x 64 out-channels.
// ---------------------------------------------------------------------------
__global__ void conv3x3_wmma(const _Float16* __restrict__ in,
                             const _Float16* __restrict__ Bp,
                             const float* __restrict__ bias,
                             float* __restrict__ y,
                             int Cpad) {
    extern __shared__ _Float16 sB[];
    const int ksteps = Cpad >> 5;
    const int nhalf  = 9 * ksteps * 4 * 512;

    // cooperative 16B-chunk copy of all B fragments into LDS
    const int tid  = threadIdx.x;
    const int nvec = nhalf >> 3;
    for (int i = tid; i < nvec; i += blockDim.x)
        ((uint4*)sB)[i] = ((const uint4*)Bp)[i];
    __syncthreads();

    const int wave = tid >> 5;
    const int lane = tid & 31;
    const int rowm = lane & 15;            // pixel row this lane feeds into A
    const int chof = (lane >> 4) << 4;     // 0 or 16: channel half within k-step

    const int tile_base = blockIdx.x * 128;
    const int p   = tile_base + wave * 16 + rowm;
    const int n   = p >> 14;
    const int rem = p & 16383;
    const int yy  = rem >> 7, xx = rem & 127;
    const long long pixbase = (long long)n * PHW + (long long)(yy + 1) * PW + (xx + 1);

    v8f acc0 = {}, acc1 = {}, acc2 = {}, acc3 = {};

    for (int tap = 0; tap < 9; ++tap) {
        const int dy = tap / 3 - 1, dx = tap % 3 - 1;
        const _Float16* abase = in + (size_t)(pixbase + (long long)dy * PW + dx) * Cpad + chof;
        for (int s = 0; s < ksteps; ++s) {
            v16h a = *(const v16h*)(abase + s * 32);
            const _Float16* bp = sB + (size_t)((tap * ksteps + s) * 4) * 512 + lane * 16;
            v16h b0 = *(const v16h*)(bp);
            v16h b1 = *(const v16h*)(bp + 512);
            v16h b2 = *(const v16h*)(bp + 1024);
            v16h b3 = *(const v16h*)(bp + 1536);
            acc0 = __builtin_amdgcn_wmma_f32_16x16x32_f16(false, a, false, b0, (short)0, acc0, false, false);
            acc1 = __builtin_amdgcn_wmma_f32_16x16x32_f16(false, a, false, b1, (short)0, acc1, false, false);
            acc2 = __builtin_amdgcn_wmma_f32_16x16x32_f16(false, a, false, b2, (short)0, acc2, false, false);
            acc3 = __builtin_amdgcn_wmma_f32_16x16x32_f16(false, a, false, b3, (short)0, acc3, false, false);
        }
    }

    // C/D layout: lanes 0-15 hold M=0..7 in v[0..7], lanes 16-31 hold M=8..15; col = lane&15
    const int col   = lane & 15;
    const int rbase = (lane >> 4) * 8;
    const float bb0 = bias[col], bb1 = bias[16 + col], bb2 = bias[32 + col], bb3 = bias[48 + col];
#pragma unroll
    for (int r = 0; r < 8; ++r) {
        size_t o = (size_t)(tile_base + wave * 16 + rbase + r) * 64;
        y[o + col]      = acc0[r] + bb0;
        y[o + 16 + col] = acc1[r] + bb1;
        y[o + 32 + col] = acc2[r] + bb2;
        y[o + 48 + col] = acc3[r] + bb3;
    }
}

// ---------------------------------------------------------------------------
// Per-channel sum / sum-of-squares over y [Np][64] (BatchNorm batch stats).
// ---------------------------------------------------------------------------
__global__ void bn_stats(const float* __restrict__ y, float* __restrict__ stats, int Np) {
    __shared__ float ssum[64], ssq[64];
    const int tid = threadIdx.x;
    if (tid < 64) { ssum[tid] = 0.f; ssq[tid] = 0.f; }
    __syncthreads();
    const int c   = tid & 63;
    const int sub = tid >> 6;
    float s = 0.f, q = 0.f;
    long long p0 = (long long)blockIdx.x * 1024;
    for (long long p = p0 + sub; p < p0 + 1024 && p < Np; p += 4) {
        float v = y[p * 64 + c];
        s += v; q += v * v;
    }
    atomicAdd(&ssum[c], s);
    atomicAdd(&ssq[c], q);
    __syncthreads();
    if (tid < 64) {
        atomicAdd(&stats[tid], ssum[tid]);
        atomicAdd(&stats[64 + tid], ssq[tid]);
    }
}

// ---------------------------------------------------------------------------
// Normalize + affine + ReLU; write next-layer f16 into padded layout
// (halo written as exact zeros each launch -> deterministic).
// ---------------------------------------------------------------------------
__global__ void bn_apply(const float* __restrict__ y, const float* __restrict__ stats,
                         const float* __restrict__ g, const float* __restrict__ be,
                         _Float16* __restrict__ out) {
    long long idx = (long long)blockIdx.x * blockDim.x + threadIdx.x;
    long long total = (long long)NPAD * 64;
    if (idx >= total) return;
    int c = (int)(idx & 63);
    long long p = idx >> 6;
    long long n = p / PHW;
    int pp = (int)(p % PHW);
    int py = pp / PW, px = pp % PW;
    float v = 0.f;
    if (py >= 1 && py <= 128 && px >= 1 && px <= 128) {
        long long src = (n * HW + (long long)(py - 1) * 128 + (px - 1)) * 64 + c;
        float cnt = (float)NPIX;
        float m   = stats[c] / cnt;
        float var = stats[64 + c] / cnt - m * m;
        float inv = rsqrtf(var + 1e-5f);
        v = fmaxf((y[src] - m) * inv * g[c] + be[c], 0.f);
    }
    out[idx] = (_Float16)v;
}

// ---------------------------------------------------------------------------
// Class prototypes: masked per-class feature sums + counts (LDS + atomics)
// acc layout: [5][65] = 64 channel sums + count.  feat is padded layout.
// ---------------------------------------------------------------------------
__global__ void proto_accum(const _Float16* __restrict__ feat, const int* __restrict__ msk,
                            const int* __restrict__ cls, float* __restrict__ acc, int Np) {
    __shared__ float sacc[5 * 65];
    const int tid = threadIdx.x;
    for (int i = tid; i < 325; i += blockDim.x) sacc[i] = 0.f;
    __syncthreads();
    for (long long p = (long long)blockIdx.x * blockDim.x + tid; p < Np;
         p += (long long)gridDim.x * blockDim.x) {
        int m = msk[p];
        int k = -1;
        for (int i = 0; i < 5; ++i) if (m == cls[i]) k = i;
        if (k >= 0) {
            atomicAdd(&sacc[k * 65 + 64], 1.f);
            const _Float16* f = feat + pad_index(p) * 64;
            for (int c = 0; c < 64; ++c) atomicAdd(&sacc[k * 65 + c], (float)f[c]);
        }
    }
    __syncthreads();
    for (int i = tid; i < 325; i += blockDim.x) atomicAdd(&acc[i], sacc[i]);
}

__global__ void finalize_protos(const float* __restrict__ acc, float* __restrict__ protos) {
    int i = blockIdx.x * blockDim.x + threadIdx.x;
    if (i >= 320) return;
    int k = i >> 6, c = i & 63;
    float cnt = acc[k * 65 + 64];
    protos[i] = acc[k * 65 + c] / fmaxf(cnt, 1.f);
}

// ---------------------------------------------------------------------------
// Squared-distance logits -> log-softmax -> masked cross-entropy accumulation
// ---------------------------------------------------------------------------
__global__ void loss_kernel(const _Float16* __restrict__ feat, const int* __restrict__ msk,
                            const int* __restrict__ cls, const float* __restrict__ protos,
                            float* __restrict__ l2, int Np) {
    __shared__ float ls[2];
    if (threadIdx.x < 2) ls[threadIdx.x] = 0.f;
    __syncthreads();
    float mysum = 0.f, mycnt = 0.f;
    for (long long p = (long long)blockIdx.x * blockDim.x + threadIdx.x; p < Np;
         p += (long long)gridDim.x * blockDim.x) {
        int m = msk[p];
        int lab = -1;
        for (int i = 0; i < 5; ++i) if (m == cls[i]) lab = i;
        if (lab < 0) continue;
        const _Float16* fp = feat + pad_index(p) * 64;
        float f[64];
        float fsq = 0.f;
        for (int c = 0; c < 64; ++c) { f[c] = (float)fp[c]; fsq += f[c] * f[c]; }
        float logits[5];
        float mx = -1e30f;
        for (int k = 0; k < 5; ++k) {
            float dot = 0.f, psq = 0.f;
            for (int c = 0; c < 64; ++c) {
                float pv = protos[k * 64 + c];
                dot += f[c] * pv;
                psq += pv * pv;
            }
            logits[k] = -(fsq + psq - 2.f * dot);
            mx = fmaxf(mx, logits[k]);
        }
        float se = 0.f;
        for (int k = 0; k < 5; ++k) se += expf(logits[k] - mx);
        float lse = mx + logf(se);
        mysum += (lse - logits[lab]);
        mycnt += 1.f;
    }
    atomicAdd(&ls[0], mysum);
    atomicAdd(&ls[1], mycnt);
    __syncthreads();
    if (threadIdx.x == 0) { atomicAdd(&l2[0], ls[0]); atomicAdd(&l2[1], ls[1]); }
}

__global__ void finalize_loss(const float* __restrict__ l2, float* __restrict__ out) {
    if (threadIdx.x == 0 && blockIdx.x == 0)
        out[0] = l2[0] / fmaxf(l2[1], 1.f);
}

__global__ void zero_f32(float* __restrict__ p, int n) {
    int i = blockIdx.x * blockDim.x + threadIdx.x;
    if (i < n) p[i] = 0.f;
}

// ---------------------------------------------------------------------------
extern "C" void kernel_launch(void* const* d_in, const int* in_sizes, int n_in,
                              void* d_out, int out_size, void* d_ws, size_t ws_size,
                              hipStream_t stream) {
    const float* xs_img = (const float*)d_in[0];
    const float* xq_img = (const float*)d_in[1];
    const int*   xs_msk = (const int*)d_in[2];
    const int*   xq_msk = (const int*)d_in[3];
    const int*   cls    = (const int*)d_in[4];
    const float* wL[3]  = {(const float*)d_in[5],  (const float*)d_in[9],  (const float*)d_in[13]};
    const float* bL[3]  = {(const float*)d_in[6],  (const float*)d_in[10], (const float*)d_in[14]};
    const float* gL[3]  = {(const float*)d_in[7],  (const float*)d_in[11], (const float*)d_in[15]};
    const float* beL[3] = {(const float*)d_in[8],  (const float*)d_in[12], (const float*)d_in[16]};

    const int Np = NPIX;

    char* ws = (char*)d_ws;
    const size_t SZ_ACT16 = ((size_t)NPAD * 64 * 2 + 255) & ~(size_t)255; // padded f16 acts
    const size_t SZ_Y32   = (size_t)Np * 64 * 4;                          // pre-BN f32
    _Float16* bufA   = (_Float16*)(ws);
    _Float16* bufB   = (_Float16*)(ws + SZ_ACT16);
    float*    ybuf   = (float*)   (ws + 2 * SZ_ACT16);
    _Float16* xsf    = (_Float16*)(ws + 2 * SZ_ACT16 + SZ_Y32);
    _Float16* Bp     = (_Float16*)(ws + 3 * SZ_ACT16 + SZ_Y32);
    float*    stats  = (float*)   (ws + 3 * SZ_ACT16 + SZ_Y32 + 131072);
    float*    acc    = stats + 128;
    float*    protos = acc + 352;
    float*    l2     = protos + 320;

    auto run_layer = [&](const _Float16* in, _Float16* out, int Ci, int Cpad, int li) {
        int ksteps = Cpad >> 5;
        int wtot = 9 * ksteps * 4 * 512;
        pack_weights<<<(wtot + 255) / 256, 256, 0, stream>>>(wL[li], Bp, Ci, Cpad);
        size_t shmem = (size_t)wtot * sizeof(_Float16);
        conv3x3_wmma<<<Np / 128, 256, shmem, stream>>>(in, Bp, bL[li], ybuf, Cpad);
        zero_f32<<<1, 128, 0, stream>>>(stats, 128);
        bn_stats<<<Np / 1024, 256, 0, stream>>>(ybuf, stats, Np);
        long long tot = (long long)NPAD * 64;
        bn_apply<<<(unsigned)((tot + 255) / 256), 256, 0, stream>>>(ybuf, stats, gL[li], beL[li], out);
    };
    auto run_encoder = [&](const float* img, _Float16* feat_out) {
        long long tin = (long long)NPAD * 32;
        pack_input<<<(unsigned)((tin + 255) / 256), 256, 0, stream>>>(img, bufA, 6, 32);
        run_layer(bufA, bufB, 6, 32, 0);
        run_layer(bufB, bufA, 64, 64, 1);
        run_layer(bufA, feat_out, 64, 64, 2);
    };

    run_encoder(xs_img, xsf);    // support features -> xsf (padded layout)
    run_encoder(xq_img, bufB);   // query features   -> bufB (padded layout)

    zero_f32<<<1, 512, 0, stream>>>(acc, 352);
    proto_accum<<<512, 256, 0, stream>>>(xsf, xs_msk, cls, acc, Np);
    finalize_protos<<<2, 256, 0, stream>>>(acc, protos);

    zero_f32<<<1, 32, 0, stream>>>(l2, 2);
    loss_kernel<<<1024, 256, 0, stream>>>(bufB, xq_msk, cls, protos, l2, Np);
    finalize_loss<<<1, 1, 0, stream>>>(l2, (float*)d_out);
}
